// SharedFieldV6_64828236366567
// MI455X (gfx1250) — matrix operand order repaired and checked by reading, not compiled, not run on hardware
//
#include <hip/hip_runtime.h>

// MI455X / gfx1250: wave32, WMMA 16x16x32 f16->f32.
typedef _Float16 f16;
typedef __attribute__((ext_vector_type(8)))  _Float16 v8h;
typedef __attribute__((ext_vector_type(16))) _Float16 v16h;
typedef __attribute__((ext_vector_type(8)))  float    v8f;

#define BB 2
#define NN 1024
#define DD 128

// ---------------------------------------------------------------------------
// Kernel 1: per 32 node rows:
//   h = X @ W_in^T          -> pair-normalize -> U (f16, row-major, B*N x 128)
//   v = (X*g) @ W_out^T     -> Vt (f16, TRANSPOSED: [B][128][1024]) so that
//                              WMMA B-fragments are contiguous 32B/lane loads.
// cos(atan2(y+e,x+e)) = (x+e)/r, sin = (y+e)/r  => no transcendentals needed.
// ---------------------------------------------------------------------------
__global__ __launch_bounds__(256) void prep_kernel(
    const float* __restrict__ X, const float* __restrict__ G,
    const float* __restrict__ Wi, const float* __restrict__ Wo,
    f16* __restrict__ U, f16* __restrict__ Vt)
{
  __shared__ float xs[32][DD];
  __shared__ float hs[32][DD];
  __shared__ float gs[32];
  const int t = threadIdx.x;
  const int rowbase = blockIdx.x * 32;               // flat row over B*N
  for (int i = t; i < 32 * DD; i += 256)
    xs[i >> 7][i & 127] = X[(size_t)rowbase * DD + i];
  if (t < 32) gs[t] = G[rowbase + t];
  __syncthreads();

  const int d  = t & 127;
  const int hh = t >> 7;                             // which 16-row half
  const float* wi = Wi + (size_t)d * DD;
  const float* wo = Wo + (size_t)d * DD;
  for (int r = 0; r < 16; ++r) {
    const int row = hh * 16 + r;
    float ah = 0.f, av = 0.f;
    for (int k = 0; k < DD; ++k) {
      const float xv = xs[row][k];
      ah = fmaf(wi[k], xv, ah);                      // h = x @ W_in^T
      av = fmaf(wo[k], xv, av);                      // pre-gate v
    }
    hs[row][d] = ah;
    const int grow = rowbase + row;
    const int b = grow >> 10, n = grow & (NN - 1);
    Vt[((size_t)b * DD + d) * NN + n] = (f16)(av * gs[row]);
  }
  __syncthreads();

  const float eps = 1e-8f;
  for (int i = 0; i < 8; ++i) {                      // 32 rows * 64 pairs
    const int idx = t + i * 256;
    const int row = idx >> 6, p = idx & 63;
    const float x0 = hs[row][2 * p]     + eps;
    const float y0 = hs[row][2 * p + 1] + eps;
    const float inv = rsqrtf(fmaf(x0, x0, y0 * y0));
    const int grow = rowbase + row;
    U[(size_t)grow * DD + 2 * p]     = (f16)(x0 * inv);
    U[(size_t)grow * DD + 2 * p + 1] = (f16)(y0 * inv);
  }
}

// ---------------------------------------------------------------------------
// Kernel 2: one wave owns a 16-row i-tile. For each 32-wide j-chunk:
//   r_dot = U_i . U_j   (4x v_wmma_f32_16x16x32_f16, K=128)
//   rnew  = clip(0.7*r_acc + 0.3*r_dot/64)         -> streamed to d_out
//   c~    = sigmoid(cond*rnew) * (i!=j) * g_col    -> f16 via LDS re-stripe
//   field += c~ @ V     (8x v_wmma, accumulating 16x128 in 8 v8f regs)
// Row gate + 1e-8-normalization applied in f32 after the GEMM:
//   field_i = g_i * acc_i / (g_i * rowsum_i + 1e-8)   (algebraically exact)
// ---------------------------------------------------------------------------
__global__ __launch_bounds__(128) void field_kernel(
    const f16* __restrict__ U, const f16* __restrict__ Vt,
    const float* __restrict__ Racc, const float* __restrict__ G,
    const float* __restrict__ Cond,
    float* __restrict__ Fout, float* __restrict__ Rout)
{
  // 40-half row stride: bank stride 20 -> 16 rows hit 16 distinct banks.
  __shared__ __align__(16) f16 cbuf[4][16][40];
  const int lane = threadIdx.x & 31;
  const int wave = threadIdx.x >> 5;
  const int iblock = blockIdx.x * 4 + wave;          // 0..127
  const int b  = iblock >> 6;
  const int i0 = (iblock & 63) << 4;                 // row base within batch
  const int lrow = lane & 15;
  const int lhi  = lane >> 4;                        // half-wave select
  const f16*   Ub  = U    + (size_t)b * NN * DD;
  const f16*   Vtb = Vt   + (size_t)b * DD * NN;
  const float* Rb  = Racc + (size_t)b * NN * NN;
  float*       Rob = Rout + (size_t)b * NN * NN;
  float cond = Cond[0];
  cond = fminf(fmaxf(cond, -5.f), 5.f);

  // A-fragments of the 16 U rows (contiguous-k order, same as B): invariant over j.
  // lane holds row (lane&15); elems 0..15 = K base..base+15,
  // base = kc*32 + (lane>=16 ? 16 : 0).
  v16h aU[4];
  {
    const f16* arow = Ub + (size_t)(i0 + lrow) * DD;
    for (int kc = 0; kc < 4; ++kc) {
      const int base = kc * 32 + lhi * 16;
      v8h lo = *(const v8h*)(arow + base);
      v8h hi = *(const v8h*)(arow + base + 8);
      v16h a = {};
      for (int e = 0; e < 8; ++e) { a[e] = lo[e]; a[e + 8] = hi[e]; }
      aU[kc] = a;
    }
  }

  v8f facc[8];
  for (int nb = 0; nb < 8; ++nb) facc[nb] = v8f{};
  float rowsum[8];
  for (int r = 0; r < 8; ++r) rowsum[r] = 0.f;

  for (int jc = 0; jc < NN; jc += 32) {
    for (int ht = 0; ht < 2; ++ht) {                 // two 16-col dot tiles
      const int jt = jc + ht * 16;
      v8f acc = v8f{};
      // B-fragment (contiguous-k order, same as A): col = lane&15, K contiguous,
      // lanes 16..31 take K+16 -> single 32B load per lane.
      const f16* bcol = Ub + (size_t)(jt + lrow) * DD + lhi * 16;
      for (int kc = 0; kc < 4; ++kc) {
        v16h bU = *(const v16h*)(bcol + kc * 32);
        acc = __builtin_amdgcn_wmma_f32_16x16x32_f16(
            false, aU[kc], false, bU, (short)0, acc, false, false);
      }
      const int gj = jt + lrow;
      const float gc = G[b * NN + gj];
      // C/D layout: slot r, lane -> (M = r + 8*lhi, Ncol = lane&15)
      for (int r = 0; r < 8; ++r) {
        const int m  = r + lhi * 8;
        const int gi = i0 + m;
        float rnew = 0.7f * Rb[(size_t)gi * NN + gj]
                   + 0.3f * (acc[r] * (1.f / 64.f));
        rnew = fminf(fmaxf(rnew, -2.f), 2.f);
        Rob[(size_t)gi * NN + gj] = rnew;
        float c = 1.f / (1.f + __expf(-cond * rnew));
        if (gi == gj) c = 0.f;                       // zero self-coupling
        c *= gc;                                     // column gate
        rowsum[r] += c;
        cbuf[wave][m][ht * 16 + lrow] = (f16)c;
      }
    }
    __syncthreads();
    // Re-stripe the 16x32 coupling tile into the A-fragment layout.
    v16h aC = {};
    {
      const f16* crow = &cbuf[wave][lrow][lhi * 16];
      v8h lo = *(const v8h*)(crow);
      v8h hi = *(const v8h*)(crow + 8);
      for (int e = 0; e < 8; ++e) { aC[e] = lo[e]; aC[e + 8] = hi[e]; }
    }
    for (int nb = 0; nb < 8; ++nb) {
      const f16* brow = Vtb + (size_t)(nb * 16 + lrow) * NN + jc + lhi * 16;
      v16h bV = *(const v16h*)(brow);
      facc[nb] = __builtin_amdgcn_wmma_f32_16x16x32_f16(
          false, aC, false, bV, (short)0, facc[nb], false, false);
    }
    __syncthreads();
  }

  // Butterfly row-sum across each 16-lane group (wave32 shfl_xor).
  for (int r = 0; r < 8; ++r) {
    float s = rowsum[r];
    s += __shfl_xor(s, 1);
    s += __shfl_xor(s, 2);
    s += __shfl_xor(s, 4);
    s += __shfl_xor(s, 8);
    rowsum[r] = s;
  }
  float scale[8];
  for (int r = 0; r < 8; ++r) {
    const int gi = i0 + r + lhi * 8;
    const float g = G[b * NN + gi];
    scale[r] = g / (g * rowsum[r] + 1e-8f);
  }
  float* Fb = Fout + (size_t)b * NN * DD;
  for (int nb = 0; nb < 8; ++nb)
    for (int r = 0; r < 8; ++r) {
      const int gi = i0 + r + lhi * 8;
      Fb[(size_t)gi * DD + nb * 16 + lrow] = facc[nb][r] * scale[r];
    }
}

// ---------------------------------------------------------------------------
extern "C" void kernel_launch(void* const* d_in, const int* in_sizes, int n_in,
                              void* d_out, int out_size, void* d_ws, size_t ws_size,
                              hipStream_t stream) {
  const float* X  = (const float*)d_in[0];   // node_states (B,N,D)
  const float* G  = (const float*)d_in[1];   // node_gates  (B,N,1)
  const float* R  = (const float*)d_in[2];   // r_acc       (B,N,N)
  const float* Wi = (const float*)d_in[3];   // W_in  (D,D)
  const float* Wo = (const float*)d_in[4];   // W_out (D,D)
  const float* C  = (const float*)d_in[5];   // conductance scalar

  float* Fout = (float*)d_out;                               // (B,N,D)
  float* Rout = Fout + (size_t)BB * NN * DD;                 // (B,N,N)

  f16* U  = (f16*)d_ws;                                      // B*N x 128
  f16* Vt = (f16*)((char*)d_ws + (size_t)BB * NN * DD * sizeof(f16)); // B x 128 x N

  prep_kernel<<<BB * NN / 32, 256, 0, stream>>>(X, G, Wi, Wo, U, Vt);
  field_kernel<<<(BB * NN / 16) / 4, 128, 0, stream>>>(U, Vt, R, G, C, Fout, Rout);
}
